// SimpleChebNet_29454885716401
// MI455X (gfx1250) — matrix operand order repaired and checked by reading, not compiled, not run on hardware
//
#include <hip/hip_runtime.h>
#include <hip/hip_bf16.h>

// ---------------------------------------------------------------------------
// SimpleChebNet on MI455X (gfx1250): 4x ChebConv(K=4) + LReLU/BN + L2norm +
// segment pooling. Dense matmuls use V_WMMA_F32_16X16X32_F16 (wave32 WMMA)
// with the K=4 Chebyshev stack fused into a single 256-deep GEMM per layer.
// Sparse L_hat propagation is one atomic scatter pass per Chebyshev order.
// ---------------------------------------------------------------------------

typedef __attribute__((ext_vector_type(16))) _Float16 v16h;
typedef __attribute__((ext_vector_type(8)))  float    v8f;

#define FLT_BIG 3.402823466e+38f

// ----------------------------- small utilities -----------------------------

__global__ void fill_f32(float* __restrict__ p, float v, int n) {
  int i = blockIdx.x * blockDim.x + threadIdx.x;
  if (i < n) p[i] = v;
}

__device__ inline void atomicMaxF(float* addr, float val) {
  unsigned int* ua = (unsigned int*)addr;
  unsigned int old = __hip_atomic_load(ua, __ATOMIC_RELAXED, __HIP_MEMORY_SCOPE_AGENT);
  while (__uint_as_float(old) < val) {
    unsigned int assumed = old;
    old = atomicCAS(ua, assumed, __float_as_uint(val));
    if (old == assumed) break;
  }
}

__device__ inline void atomicMinF(float* addr, float val) {
  unsigned int* ua = (unsigned int*)addr;
  unsigned int old = __hip_atomic_load(ua, __ATOMIC_RELAXED, __HIP_MEMORY_SCOPE_AGENT);
  while (__uint_as_float(old) > val) {
    unsigned int assumed = old;
    old = atomicCAS(ua, assumed, __float_as_uint(val));
    if (old == assumed) break;
  }
}

// --------------------------- graph preprocessing ---------------------------

__global__ void degree_kernel(const int* __restrict__ src, float* __restrict__ deg, int E) {
  int e = blockIdx.x * blockDim.x + threadIdx.x;
  if (e < E) atomicAdd(&deg[src[e]], 1.0f);
}

__global__ void dinv_kernel(float* __restrict__ deg, int N) {
  int n = blockIdx.x * blockDim.x + threadIdx.x;
  if (n < N) {
    float d = deg[n];
    deg[n] = (d > 0.0f) ? rsqrtf(fmaxf(d, 1.0f)) : 0.0f;
  }
}

__global__ void edgew_kernel(const int* __restrict__ src, const int* __restrict__ dst,
                             const float* __restrict__ dinv, float* __restrict__ ew, int E) {
  int e = blockIdx.x * blockDim.x + threadIdx.x;
  if (e < E) ew[e] = -(dinv[src[e]] * dinv[dst[e]]);
}

// -------- weight prep: [K, in, out] f32 -> zero-padded, K-concat, ----------
// -------- column-major f16 (Wt[col*Kcat + k*slotW + i])            ----------

__global__ void prep_w_kernel(const float* __restrict__ W, _Float16* __restrict__ Wt,
                              int inDim, int slotW, int ncol) {
  int Kcat = 4 * slotW;
  int idx = blockIdx.x * blockDim.x + threadIdx.x;
  if (idx >= Kcat * ncol) return;
  int col = idx % ncol;
  int kc  = idx / ncol;
  int k = kc / slotW, i = kc % slotW;
  float v = (i < inDim) ? W[((size_t)k * inDim + i) * ncol + col] : 0.0f;
  Wt[(size_t)col * Kcat + kc] = (_Float16)v;
}

// ------------------------- Chebyshev recurrence ----------------------------
// TX buffer: [N, 4*slotW], slot k at column offset k*slotW.

__global__ void copy_x_kernel(const float* __restrict__ x, float* __restrict__ TX, int N) {
  int idx = blockIdx.x * blockDim.x + threadIdx.x;
  if (idx >= N * 3) return;
  int n = idx / 3, i = idx % 3;
  TX[(size_t)n * 32 + i] = x[idx];
}

// row[soutOff + f] = coef * row[sinOff + f]   (coef==0 -> pure zero seed)
__global__ void cheb_seed_kernel(float* __restrict__ TX, int stride, int F,
                                 int sinOff, int soutOff, float coef, int N) {
  int idx = blockIdx.x * blockDim.x + threadIdx.x;
  if (idx >= N * F) return;
  int n = idx / F, f = idx % F;
  float* row = TX + (size_t)n * stride;
  row[soutOff + f] = (coef == 0.0f) ? 0.0f : coef * row[sinOff + f];
}

// out[dst] += scale * ew[e] * in[src]  (slot-offset pointers passed in)
__global__ void cheb_prop_kernel(const int* __restrict__ src, const int* __restrict__ dst,
                                 const float* __restrict__ ew, float scale,
                                 const float* __restrict__ TXin, float* __restrict__ TXout,
                                 int stride, int F, int E) {
  int e = blockIdx.x * blockDim.x + threadIdx.x;
  if (e >= E) return;
  float w = scale * ew[e];
  const float* in = TXin + (size_t)src[e] * stride;
  float*       out = TXout + (size_t)dst[e] * stride;
  for (int f = 0; f < F; ++f) atomicAdd(&out[f], w * in[f]);
}

// ------------------------------ WMMA GEMM ----------------------------------
// One wave computes a 16 x (NT*16) tile of H = TX(N x Kcat) @ Wt + bias.
// A fragment: lane<16: row=lane, K = kc+{0..15};
//             lane>=16: row=lane-16, K = kc+{16..31}.
// B fragment: contiguous 16 halves of column-major f16 weights.
// C/D: col = lane&15, row = r + (lane>=16 ? 8 : 0).

template <int NT, bool LRELU>
__global__ void cheb_gemm_wmma(const float* __restrict__ TX, int Kcat,
                               const _Float16* __restrict__ Wt,
                               const float* __restrict__ bias,
                               float* __restrict__ H, int nRows) {
  const int ncol = NT * 16;
  int lane = threadIdx.x & 31;
  int wave = blockIdx.x * (blockDim.x >> 5) + (threadIdx.x >> 5);
  int rowbase = wave << 4;
  if (rowbase >= nRows) return;  // wave-uniform

  int rowA  = rowbase + (lane & 15);
  int koffA = (lane & 16) ? 16 : 0;
  int colB  = lane & 15;
  int koffB = (lane & 16) ? 16 : 0;

  v8f zero = {};
  v8f acc[NT];
#pragma unroll
  for (int t = 0; t < NT; ++t) acc[t] = zero;

  const float* arow = TX + (size_t)rowA * Kcat;
  for (int kc = 0; kc < Kcat; kc += 32) {
    const float* ap = arow + kc + koffA;
    v16h a;
#pragma unroll
    for (int i = 0; i < 8; ++i) a[i] = (_Float16)ap[i];
#pragma unroll
    for (int i = 0; i < 8; ++i) a[8 + i] = (_Float16)ap[8 + i];
#pragma unroll
    for (int t = 0; t < NT; ++t) {
      const _Float16* bp = Wt + (size_t)(t * 16 + colB) * Kcat + kc + koffB;
      v16h b = *(const v16h*)bp;
      acc[t] = __builtin_amdgcn_wmma_f32_16x16x32_f16(
          false, a, false, b, (short)0, acc[t], false, false);
    }
  }

  int rowC = rowbase + ((lane & 16) ? 8 : 0);
#pragma unroll
  for (int t = 0; t < NT; ++t) {
    int col = t * 16 + (lane & 15);
    float bv = bias[col];
#pragma unroll
    for (int r = 0; r < 8; ++r) {
      float v = acc[t][r] + bv;
      if (LRELU) v = (v > 0.0f) ? v : 0.01f * v;
      H[(size_t)(rowC + r) * ncol + col] = v;
    }
  }
}

// ------------------------------- BatchNorm ---------------------------------
// H already has LeakyReLU applied; stats = [sum(64), sumsq(64)].

__global__ void bn_stats_kernel(const float* __restrict__ H, int nRows,
                                float* __restrict__ stats) {
  __shared__ float ssum[256], ssq[256];
  int tid = threadIdx.x;        // 256 threads
  int c = tid & 63;
  int ro = tid >> 6;            // 0..3
  float s = 0.0f, q = 0.0f;
  for (int n = blockIdx.x * 4 + ro; n < nRows; n += gridDim.x * 4) {
    float v = H[(size_t)n * 64 + c];
    s += v; q += v * v;
  }
  ssum[tid] = s; ssq[tid] = q;
  __syncthreads();
  if (ro == 0) {
    s = ssum[c] + ssum[64 + c] + ssum[128 + c] + ssum[192 + c];
    q = ssq[c] + ssq[64 + c] + ssq[128 + c] + ssq[192 + c];
    atomicAdd(&stats[c], s);
    atomicAdd(&stats[64 + c], q);
  }
}

// (h - mu) * rsqrt(var + eps) * g + bt  -> TX slot0 of next layer (stride 256)
__global__ void bn_apply_kernel(const float* __restrict__ H, const float* __restrict__ stats,
                                const float* __restrict__ g, const float* __restrict__ bt,
                                float* __restrict__ TX, int nRows) {
  int idx = blockIdx.x * blockDim.x + threadIdx.x;
  if (idx >= nRows * 64) return;
  int c = idx & 63, n = idx >> 6;
  float invN = 1.0f / (float)nRows;
  float mu  = stats[c] * invN;
  float var = stats[64 + c] * invN - mu * mu;
  float v = (H[idx] - mu) * rsqrtf(var + 1e-5f) * g[c] + bt[c];
  TX[(size_t)n * 256 + c] = v;
}

// ------------------------- L2 row-normalize (C=32) -------------------------

__global__ void row_l2norm_kernel(float* __restrict__ H, int N) {
  int n = blockIdx.x * blockDim.x + threadIdx.x;
  if (n >= N) return;
  float* row = H + (size_t)n * 32;
  float s = 0.0f;
  for (int c = 0; c < 32; ++c) { float v = row[c]; s += v * v; }
  float inv = 1.0f / fmaxf(sqrtf(s), 1e-12f);
  for (int c = 0; c < 32; ++c) row[c] *= inv;
}

// ------------------------------- pooling -----------------------------------

__global__ void pool_init_kernel(float* __restrict__ cnt, float* __restrict__ psum,
                                 float* __restrict__ pmax, float* __restrict__ pmin) {
  int idx = blockIdx.x * blockDim.x + threadIdx.x;
  if (idx < 64 * 32) {
    psum[idx] = 0.0f;
    pmax[idx] = -FLT_BIG;
    pmin[idx] = FLT_BIG;
    if (idx < 64) cnt[idx] = 0.0f;
  }
}

__global__ void pool_acc_kernel(const float* __restrict__ H, const int* __restrict__ batch,
                                float* __restrict__ cnt, float* __restrict__ psum,
                                float* __restrict__ pmax, float* __restrict__ pmin, int N) {
  int n = blockIdx.x * blockDim.x + threadIdx.x;
  if (n >= N) return;
  int g = batch[n];
  atomicAdd(&cnt[g], 1.0f);
  const float* row = H + (size_t)n * 32;
  for (int c = 0; c < 32; ++c) {
    float v = row[c];
    atomicAdd(&psum[g * 32 + c], v);
    atomicMaxF(&pmax[g * 32 + c], v);
    atomicMinF(&pmin[g * 32 + c], v);
  }
}

__global__ void pool_fin_kernel(const float* __restrict__ cnt, const float* __restrict__ psum,
                                const float* __restrict__ pmax, const float* __restrict__ pmin,
                                float* __restrict__ out) {
  int idx = blockIdx.x * blockDim.x + threadIdx.x;
  if (idx >= 64 * 32) return;
  int g = idx / 32, c = idx % 32;
  float cn = fmaxf(cnt[g], 1.0f);
  out[(size_t)g * 128 + c]      = psum[idx] / cn;   // mean
  out[(size_t)g * 128 + 32 + c] = pmax[idx];        // max
  out[(size_t)g * 128 + 64 + c] = pmin[idx];        // min
  out[(size_t)g * 128 + 96 + c] = psum[idx];        // sum
}

// ------------------------------- launcher ----------------------------------

static inline char* carve(char*& p, size_t bytes) {
  char* r = p;
  p += (bytes + 255) & ~(size_t)255;
  return r;
}

static void run_cheb_props(const int* src, const int* dst, const float* ew, float* TX,
                           int stride, int F, int slotW, int N, int E, hipStream_t s) {
  int sb = (N * F + 255) / 256;        // seed blocks
  int pb = (E + 255) / 256;            // prop blocks
  int s0 = 0, s1 = slotW, s2 = 2 * slotW, s3 = 3 * slotW;
  // Tx1 = prop(Tx0)
  cheb_seed_kernel<<<sb, 256, 0, s>>>(TX, stride, F, s0, s1, 0.0f, N);
  cheb_prop_kernel<<<pb, 256, 0, s>>>(src, dst, ew, 1.0f, TX + s0, TX + s1, stride, F, E);
  // Tx2 = 2*prop(Tx1) - Tx0
  cheb_seed_kernel<<<sb, 256, 0, s>>>(TX, stride, F, s0, s2, -1.0f, N);
  cheb_prop_kernel<<<pb, 256, 0, s>>>(src, dst, ew, 2.0f, TX + s1, TX + s2, stride, F, E);
  // Tx3 = 2*prop(Tx2) - Tx1
  cheb_seed_kernel<<<sb, 256, 0, s>>>(TX, stride, F, s1, s3, -1.0f, N);
  cheb_prop_kernel<<<pb, 256, 0, s>>>(src, dst, ew, 2.0f, TX + s2, TX + s3, stride, F, E);
}

extern "C" void kernel_launch(void* const* d_in, const int* in_sizes, int n_in,
                              void* d_out, int out_size, void* d_ws, size_t ws_size,
                              hipStream_t stream) {
  (void)n_in; (void)out_size; (void)ws_size;

  const float* x     = (const float*)d_in[0];
  const int*   ei    = (const int*)d_in[1];   // [2, E]
  const int*   batch = (const int*)d_in[2];
  const float* W1  = (const float*)d_in[3];  const float* bc1 = (const float*)d_in[4];
  const float* W2  = (const float*)d_in[5];  const float* bc2 = (const float*)d_in[6];
  const float* W3  = (const float*)d_in[7];  const float* bc3 = (const float*)d_in[8];
  const float* W4  = (const float*)d_in[9];  const float* bc4 = (const float*)d_in[10];
  const float* g1  = (const float*)d_in[11]; const float* bt1 = (const float*)d_in[12];
  const float* g2  = (const float*)d_in[13]; const float* bt2 = (const float*)d_in[14];
  const float* g3  = (const float*)d_in[15]; const float* bt3 = (const float*)d_in[16];

  const int N = in_sizes[0] / 3;
  const int E = in_sizes[1] / 2;
  const int* src = ei;
  const int* dst = ei + E;

  // ---- workspace carve ----
  char* p = (char*)d_ws;
  float* ew    = (float*)carve(p, (size_t)E * 4);
  float* deg   = (float*)carve(p, (size_t)N * 4);       // becomes dinv in place
  float* TX    = (float*)carve(p, (size_t)N * 256 * 4); // Chebyshev stack [N, 4*64]
  float* H     = (float*)carve(p, (size_t)N * 64 * 4);  // layer output
  float* stats = (float*)carve(p, 128 * 4);
  float* cnt   = (float*)carve(p, 64 * 4);
  float* psum  = (float*)carve(p, 64 * 32 * 4);
  float* pmax  = (float*)carve(p, 64 * 32 * 4);
  float* pmin  = (float*)carve(p, 64 * 32 * 4);
  _Float16* Wt1 = (_Float16*)carve(p, (size_t)32 * 64 * 2);
  _Float16* Wt2 = (_Float16*)carve(p, (size_t)256 * 64 * 2);
  _Float16* Wt3 = (_Float16*)carve(p, (size_t)256 * 64 * 2);
  _Float16* Wt4 = (_Float16*)carve(p, (size_t)256 * 32 * 2);

  const int nb = 256;

  // ---- Laplacian edge weights ----
  fill_f32<<<(N + nb - 1) / nb, nb, 0, stream>>>(deg, 0.0f, N);
  degree_kernel<<<(E + nb - 1) / nb, nb, 0, stream>>>(src, deg, E);
  dinv_kernel<<<(N + nb - 1) / nb, nb, 0, stream>>>(deg, N);
  edgew_kernel<<<(E + nb - 1) / nb, nb, 0, stream>>>(src, dst, deg, ew, E);

  // ---- weight prep (f32 [K,in,out] -> f16 column-major K-concat) ----
  prep_w_kernel<<<(32 * 64 + nb - 1) / nb, nb, 0, stream>>>(W1, Wt1, 3, 8, 64);
  prep_w_kernel<<<(256 * 64 + nb - 1) / nb, nb, 0, stream>>>(W2, Wt2, 64, 64, 64);
  prep_w_kernel<<<(256 * 64 + nb - 1) / nb, nb, 0, stream>>>(W3, Wt3, 64, 64, 64);
  prep_w_kernel<<<(256 * 32 + nb - 1) / nb, nb, 0, stream>>>(W4, Wt4, 64, 64, 32);

  // GEMM launch config: 1 wave per 16 rows, 2 waves per block
  int waves = (N + 15) / 16;
  int gemmBlocks = (waves + 1) / 2;
  int gemmThreads = 64;

  // ---- layer 1 (in=3 zero-padded to slotW=8, Kcat=32) ----
  fill_f32<<<(N * 32 + nb - 1) / nb, nb, 0, stream>>>(TX, 0.0f, N * 32);
  copy_x_kernel<<<(N * 3 + nb - 1) / nb, nb, 0, stream>>>(x, TX, N);
  run_cheb_props(src, dst, ew, TX, 32, 3, 8, N, E, stream);
  cheb_gemm_wmma<4, true><<<gemmBlocks, gemmThreads, 0, stream>>>(TX, 32, Wt1, bc1, H, N);
  fill_f32<<<1, 128, 0, stream>>>(stats, 0.0f, 128);
  bn_stats_kernel<<<512, 256, 0, stream>>>(H, N, stats);
  bn_apply_kernel<<<(N * 64 + nb - 1) / nb, nb, 0, stream>>>(H, stats, g1, bt1, TX, N);

  // ---- layer 2 ----
  run_cheb_props(src, dst, ew, TX, 256, 64, 64, N, E, stream);
  cheb_gemm_wmma<4, true><<<gemmBlocks, gemmThreads, 0, stream>>>(TX, 256, Wt2, bc2, H, N);
  fill_f32<<<1, 128, 0, stream>>>(stats, 0.0f, 128);
  bn_stats_kernel<<<512, 256, 0, stream>>>(H, N, stats);
  bn_apply_kernel<<<(N * 64 + nb - 1) / nb, nb, 0, stream>>>(H, stats, g2, bt2, TX, N);

  // ---- layer 3 ----
  run_cheb_props(src, dst, ew, TX, 256, 64, 64, N, E, stream);
  cheb_gemm_wmma<4, true><<<gemmBlocks, gemmThreads, 0, stream>>>(TX, 256, Wt3, bc3, H, N);
  fill_f32<<<1, 128, 0, stream>>>(stats, 0.0f, 128);
  bn_stats_kernel<<<512, 256, 0, stream>>>(H, N, stats);
  bn_apply_kernel<<<(N * 64 + nb - 1) / nb, nb, 0, stream>>>(H, stats, g3, bt3, TX, N);

  // ---- layer 4 (out=32, no activation/BN) ----
  run_cheb_props(src, dst, ew, TX, 256, 64, 64, N, E, stream);
  cheb_gemm_wmma<2, false><<<gemmBlocks, gemmThreads, 0, stream>>>(TX, 256, Wt4, bc4, H, N);

  // ---- L2 row-normalize + pooling ----
  row_l2norm_kernel<<<(N + nb - 1) / nb, nb, 0, stream>>>(H, N);
  pool_init_kernel<<<(64 * 32 + nb - 1) / nb, nb, 0, stream>>>(cnt, psum, pmax, pmin);
  pool_acc_kernel<<<(N + nb - 1) / nb, nb, 0, stream>>>(H, batch, cnt, psum, pmax, pmin, N);
  pool_fin_kernel<<<(64 * 32 + nb - 1) / nb, nb, 0, stream>>>(cnt, psum, pmax, pmin,
                                                              (float*)d_out);
}
